// SwinBlock_12738873000256
// MI455X (gfx1250) — matrix operand order repaired and checked by reading, not compiled, not run on hardware
//
#include <hip/hip_runtime.h>
#include <hip/hip_bf16.h>

// ---------------------------------------------------------------------------
// Swin block on gfx1250 (MI455X): all GEMMs via v_wmma_f32_16x16x32_f16.
//   B=32, H=W=56, C=256, NH=8, HD=32, WS=7, SS=3, L=49 (padded to 64), F=1024
// Per-wave row-tile is constant => A-fragments hoisted into registers; B
// fragments batched so WMMA chains run on partial loadcnt waits.  Wave index
// is readfirstlane'd so tile control flow / epilogue selection is scalar.
// ---------------------------------------------------------------------------

typedef __attribute__((ext_vector_type(16))) _Float16 v16h;
typedef __attribute__((ext_vector_type(8)))  float    v8f;

#define WMMA_F16(a, b, c) \
    __builtin_amdgcn_wmma_f32_16x16x32_f16(false, (a), false, (b), (short)0, (c), false, false)

// ---- fragment loaders (contiguous k = 16*(lane/16) + i, same for A and B) ----------
// A: 16x32 f16 tile, row-major, ld in halves.
//   lanes 0-15 : row=lane,    K = 0..15
//   lanes 16-31: row=lane-16, K = 16..31
__device__ __forceinline__ v16h ldsA(const _Float16* base, int ld) {
    const int lane = threadIdx.x & 31;
    const int row  = lane & 15;
    const int khi  = (lane >> 4) << 4;          // 0 or 16
    const _Float16* r0 = base + row * ld + khi;
    v16h a;
#pragma unroll
    for (int i = 0; i < 8; ++i) a[i] = r0[i];
#pragma unroll
    for (int i = 0; i < 8; ++i) a[8 + i] = r0[8 + i];
    return a;
}

// B: 32x16 (KxN) tile stored as [N][K] (column n contiguous in K), ld in halves.
//   lanes 0-15 : n=lane,    K = 0..15   |  lanes 16-31: n=lane-16, K = 16..31
__device__ __forceinline__ v16h ldB(const _Float16* base, int ld) {
    const int lane = threadIdx.x & 31;
    const int n    = lane & 15;
    const int k0   = (lane >> 4) << 4;          // 0 or 16
    const _Float16* r = base + n * ld + k0;
    v16h b;
#pragma unroll
    for (int i = 0; i < 16; ++i) b[i] = r[i];
    return b;
}

// ---------------------------------------------------------------------------
// Kernel 0: weight precision/layout prep into workspace (all f16, [N][K]).
//   ws16 + 0      : qkv_w  [768][256]
//   ws16 + 196608 : proj_w [256][256]
//   ws16 + 262144 : w1^T   [1024][256]
//   ws16 + 524288 : w2^T   [256][1024]
// ---------------------------------------------------------------------------
__global__ void swin_prep_weights(const float* __restrict__ qkv_w,
                                  const float* __restrict__ proj_w,
                                  const float* __restrict__ w1,
                                  const float* __restrict__ w2,
                                  _Float16* __restrict__ ws16) {
    const int tid    = blockIdx.x * blockDim.x + threadIdx.x;
    const int stride = gridDim.x * blockDim.x;
    for (int i = tid; i < 768 * 256; i += stride) ws16[i] = (_Float16)qkv_w[i];
    _Float16* pj = ws16 + 196608;
    for (int i = tid; i < 256 * 256; i += stride) pj[i] = (_Float16)proj_w[i];
    _Float16* t1 = ws16 + 262144;                     // [f][c] <- w1[c][f]
    for (int i = tid; i < 1024 * 256; i += stride) {
        int f = i >> 8, c = i & 255;
        t1[i] = (_Float16)w1[c * 1024 + f];
    }
    _Float16* t2 = ws16 + 524288;                     // [c][f] <- w2[f][c]
    for (int i = tid; i < 256 * 1024; i += stride) {
        int c = i >> 10, f = i & 1023;
        t2[i] = (_Float16)w2[f * 256 + c];
    }
}

// ---------------------------------------------------------------------------
// Kernel 1: windowed attention + proj + residual LayerNorm.  One block per
// window (2048 blocks, 256 threads = 8 waves).  Dynamic LDS (bytes):
//   [0)      xh  : 64x264 f16   window activations, later reused for O
//   [33792)  qh  : 64x264 f16   Q (pre-scaled)        } aliased by abuf
//   [67584)  kh  : 64x264 f16   K                     } (64x260 f32) in proj
//   [101376) vt  : 256x72 f16   V^T
//   [138240) sb  : 8 x 64x66 f16  per-wave scores/probs
//   [205824) rpb : 169x8 f32
//   [211232) reg : 64 int       shift-mask region labels
// total 211488 bytes (CDNA5 WGP LDS = 320KB)
// ---------------------------------------------------------------------------
#define LDQ 264
#define LDV 72
#define LDSR 66
#define LDA 260

__global__ __launch_bounds__(256) void swin_attn_kernel(
        const float* __restrict__ x,
        const _Float16* __restrict__ qkvw,   // [768][256] f16
        const float* __restrict__ qkvb,
        const _Float16* __restrict__ projw,  // [256][256] f16
        const float* __restrict__ projb,
        const float* __restrict__ rpb,       // [169][8]
        const float* __restrict__ gamma1,
        const float* __restrict__ beta1,
        float* __restrict__ y) {
    extern __shared__ char smem[];
    _Float16* xh   = (_Float16*)(smem);             // also O (attn output)
    _Float16* qh   = (_Float16*)(smem + 33792);
    _Float16* kh   = (_Float16*)(smem + 67584);
    _Float16* vt   = (_Float16*)(smem + 101376);
    _Float16* sb   = (_Float16*)(smem + 138240);
    float*    abuf = (float*)(smem + 33792);        // aliases qh/kh (proj out)
    float*    rpbs = (float*)(smem + 205824);
    int*      regn = (int*)(smem + 211232);

    const int tid  = threadIdx.x;
    const int wave = __builtin_amdgcn_readfirstlane(tid >> 5);   // scalar
    const int lane = tid & 31;
    const int blk  = blockIdx.x;
    const int b    = blk >> 6;
    const int win  = blk & 63;
    const int wr   = win >> 3;
    const int wc   = win & 7;

    // ---- phase 0: stage rpb, region labels, shifted window (f32->f16) ----
    for (int i = tid; i < 169 * 8; i += 256) rpbs[i] = rpb[i];
    if (tid < 64) {
        int rg = 0;
        if (tid < 49) {
            int i = tid / 7, j = tid % 7;
            int h = wr * 7 + i, w = wc * 7 + j;
            int fh = (h < 49) ? 0 : ((h < 53) ? 1 : 2);
            int fw = (w < 49) ? 0 : ((w < 53) ? 1 : 2);
            rg = fh * 3 + fw;
        }
        regn[tid] = rg;
    }
    for (int idx = tid; idx < 64 * 256; idx += 256) {
        int m = idx >> 8, c = idx & 255;
        float v = 0.f;
        if (m < 49) {
            int i = m / 7, j = m % 7;
            int h = (wr * 7 + i + 3) % 56;               // cyclic shift -3
            int w = (wc * 7 + j + 3) % 56;
            v = x[(((size_t)b * 56 + h) * 56 + w) * 256 + c];
        }
        xh[m * LDQ + c] = (_Float16)v;
    }
    __syncthreads();

    // ---- phase 1: QKV = xh[64x256] * qkvw^T -> q,k (row), v (transposed) --
    const float qscale = 0.17677669529663687f;          // HD^-0.5
    {
        const int mt = wave & 3;
        v16h af[8];
#pragma unroll
        for (int kt = 0; kt < 8; ++kt) af[kt] = ldsA(xh + mt * 16 * LDQ + kt * 32, LDQ);
        for (int i = 0; i < 24; ++i) {
            const int nt = (wave >> 2) + 2 * i;         // 0..47 (wave-uniform)
            __builtin_prefetch(qkvw + (nt + 2) * 16 * 256, 0, 1);
            v16h bf[8];
#pragma unroll
            for (int kt = 0; kt < 8; ++kt) bf[kt] = ldB(qkvw + nt * 16 * 256 + kt * 32, 256);
            v8f c = {};
#pragma unroll
            for (int kt = 0; kt < 8; ++kt) c = WMMA_F16(af[kt], bf[kt], c);
            const int n16 = lane & 15;
            const int mb  = (lane >> 4) << 3;
            const int n   = nt * 16 + n16;              // global qkv channel
            const float bias = qkvb[n];
            if (nt < 16) {                               // Q segment (scalar br)
#pragma unroll
                for (int r = 0; r < 8; ++r) {
                    int m = mt * 16 + mb + r;
                    qh[m * LDQ + n] = (_Float16)((c[r] + bias) * qscale);
                }
            } else if (nt < 32) {                        // K segment
#pragma unroll
                for (int r = 0; r < 8; ++r) {
                    int m = mt * 16 + mb + r;
                    kh[m * LDQ + (n - 256)] = (_Float16)(c[r] + bias);
                }
            } else {                                     // V segment, transposed
#pragma unroll
                for (int r = 0; r < 8; ++r) {
                    int m = mt * 16 + mb + r;
                    vt[(n - 512) * LDV + m] = (_Float16)(c[r] + bias);
                }
            }
        }
    }
    __syncthreads();

    // ---- phase 2: per-head attention (wave == head) -----------------------
    _Float16* sw = sb + wave * 64 * LDSR;
    for (int mt = 0; mt < 4; ++mt) {
        v16h a = ldsA(qh + mt * 16 * LDQ + wave * 32, LDQ);     // K-dim = 32
        for (int nt = 0; nt < 4; ++nt) {
            v16h bm = ldB(kh + nt * 16 * LDQ + wave * 32, LDQ);
            v8f cz = {};
            v8f c = WMMA_F16(a, bm, cz);
            int n  = nt * 16 + (lane & 15);
            int mb = (lane >> 4) << 3;
#pragma unroll
            for (int r = 0; r < 8; ++r) {
                int m = mt * 16 + mb + r;
                float v = 0.f;
                if (m < 49 && n < 49) {
                    int mi = m / 7, mj = m % 7, ni = n / 7, nj = n % 7;
                    int ridx = (mi - ni + 6) * 13 + (mj - nj + 6);
                    v = c[r] + rpbs[ridx * 8 + wave];
                    if (regn[m] != regn[n]) v -= 100.f;
                }
                sw[m * LDSR + n] = (_Float16)v;
            }
        }
    }
    __syncthreads();

    // softmax over valid 49 columns; zero the padding (2 rows per lane)
    for (int rr = 0; rr < 2; ++rr) {
        int m = lane + rr * 32;
        if (m < 49) {
            float mx = -1e30f;
            for (int j = 0; j < 49; ++j) mx = fmaxf(mx, (float)sw[m * LDSR + j]);
            float s = 0.f;
            for (int j = 0; j < 49; ++j) s += __expf((float)sw[m * LDSR + j] - mx);
            float inv = 1.f / s;
            for (int j = 0; j < 49; ++j) {
                float p = __expf((float)sw[m * LDSR + j] - mx) * inv;
                sw[m * LDSR + j] = (_Float16)p;
            }
            for (int j = 49; j < 64; ++j) sw[m * LDSR + j] = (_Float16)0.f;
        } else {
            for (int j = 0; j < 64; ++j) sw[m * LDSR + j] = (_Float16)0.f;
        }
    }
    __syncthreads();

    // O = P[64x64] * V[64x32]  (B operand = V^T rows, [N=d][K=j])
    for (int mt = 0; mt < 4; ++mt) {
        v16h a0 = ldsA(sw + mt * 16 * LDSR, LDSR);
        v16h a1 = ldsA(sw + mt * 16 * LDSR + 32, LDSR);
        for (int nt = 0; nt < 2; ++nt) {
            const _Float16* vb = vt + (wave * 32 + nt * 16) * LDV;
            v16h b0 = ldB(vb, LDV);
            v16h b1 = ldB(vb + 32, LDV);
            v8f c = {};
            c = WMMA_F16(a0, b0, c);
            c = WMMA_F16(a1, b1, c);
            int n  = lane & 15;
            int mb = (lane >> 4) << 3;
#pragma unroll
            for (int r = 0; r < 8; ++r) {
                int m = mt * 16 + mb + r;
                xh[m * LDQ + wave * 32 + nt * 16 + n] = (_Float16)c[r]; // O->xh
            }
        }
    }
    __syncthreads();

    // ---- phase 3: proj = O[64x256] * projw^T  (out f32 into abuf) ---------
    {
        const int mt = wave & 3;
        v16h af[8];
#pragma unroll
        for (int kt = 0; kt < 8; ++kt) af[kt] = ldsA(xh + mt * 16 * LDQ + kt * 32, LDQ);
        for (int i = 0; i < 8; ++i) {
            const int nt = (wave >> 2) + 2 * i;          // 0..15
            v16h bf[8];
#pragma unroll
            for (int kt = 0; kt < 8; ++kt) bf[kt] = ldB(projw + nt * 16 * 256 + kt * 32, 256);
            v8f c = {};
#pragma unroll
            for (int kt = 0; kt < 8; ++kt) c = WMMA_F16(af[kt], bf[kt], c);
            int n  = nt * 16 + (lane & 15);
            int mb = (lane >> 4) << 3;
            float bias = projb[n];
#pragma unroll
            for (int r = 0; r < 8; ++r) abuf[(mt * 16 + mb + r) * LDA + n] = c[r] + bias;
        }
    }
    __syncthreads();

    // ---- phase 4: y = x + LayerNorm(proj)  (4 lanes per token row) --------
    {
        int m = tid >> 2, quarter = tid & 3;
        float s = 0.f, sq = 0.f;
        for (int i = 0; i < 64; ++i) {
            float v = abuf[m * LDA + quarter * 64 + i];
            s += v; sq += v * v;
        }
        s  += __shfl_xor(s, 1);  s  += __shfl_xor(s, 2);
        sq += __shfl_xor(sq, 1); sq += __shfl_xor(sq, 2);
        float mu  = s * (1.f / 256.f);
        float var = sq * (1.f / 256.f) - mu * mu;
        float rs  = rsqrtf(var + 1e-5f);
        if (m < 49) {
            int i = m / 7, j = m % 7;
            int h = (wr * 7 + i + 3) % 56;      // reverse shift == load coord
            int w = (wc * 7 + j + 3) % 56;
            size_t base = (((size_t)b * 56 + h) * 56 + w) * 256;
            for (int q2 = 0; q2 < 64; ++q2) {
                int c = quarter * 64 + q2;
                float a = abuf[m * LDA + c];
                y[base + c] = x[base + c] + (a - mu) * rs * gamma1[c] + beta1[c];
            }
        }
    }
}

// ---------------------------------------------------------------------------
// Kernel 2: MLP (fc1 + exact GeLU + fc2) + residual LayerNorm, in-place on y.
// 64 token rows per block (1568 blocks).  LDS: yh 33792 + hh 132096 +
// mbuf 66560 = 232448 bytes.
// ---------------------------------------------------------------------------
#define LDY 264
#define LDH 1032
#define LDM 260

__global__ __launch_bounds__(256) void swin_mlp_kernel(
        const _Float16* __restrict__ w1t,    // [1024][256] f16
        const float* __restrict__ b1,
        const _Float16* __restrict__ w2t,    // [256][1024] f16
        const float* __restrict__ b2,
        const float* __restrict__ gamma2,
        const float* __restrict__ beta2,
        float* __restrict__ y) {
    extern __shared__ char smem[];
    _Float16* yh   = (_Float16*)(smem);
    _Float16* hh   = (_Float16*)(smem + 33792);
    float*    mbuf = (float*)(smem + 165888);

    const int tid  = threadIdx.x;
    const int wave = __builtin_amdgcn_readfirstlane(tid >> 5);   // scalar
    const int lane = tid & 31;
    const size_t row0 = (size_t)blockIdx.x * 64;

    for (int idx = tid; idx < 64 * 256; idx += 256) {
        int m = idx >> 8, c = idx & 255;
        yh[m * LDY + c] = (_Float16)y[(row0 + m) * 256 + c];
    }
    __syncthreads();

    // fc1 + GeLU: [64x256] x [256x1024]; A-fragments hoisted (mt fixed/wave)
    {
        const int mt = wave & 3;
        v16h af[8];
#pragma unroll
        for (int kt = 0; kt < 8; ++kt) af[kt] = ldsA(yh + mt * 16 * LDY + kt * 32, LDY);
        for (int i = 0; i < 32; ++i) {
            const int nt = (wave >> 2) + 2 * i;          // 0..63
            __builtin_prefetch(w1t + (nt + 2) * 16 * 256, 0, 1);
            v16h bf[8];
#pragma unroll
            for (int kt = 0; kt < 8; ++kt) bf[kt] = ldB(w1t + nt * 16 * 256 + kt * 32, 256);
            v8f c = {};
#pragma unroll
            for (int kt = 0; kt < 8; ++kt) c = WMMA_F16(af[kt], bf[kt], c);
            int n  = nt * 16 + (lane & 15);
            int mb = (lane >> 4) << 3;
            float bias = b1[n];
#pragma unroll
            for (int r = 0; r < 8; ++r) {
                float v = c[r] + bias;
                float g = 0.5f * v * (1.f + erff(v * 0.70710678118654752f));
                hh[(mt * 16 + mb + r) * LDH + n] = (_Float16)g;
            }
        }
    }
    __syncthreads();

    // fc2: [64x1024] x [1024x256].  K=1024: keep the wave's 8 accumulators
    // live and load each A-fragment once per kt (amortized over 8 WMMAs).
    {
        const int mt  = wave & 3;
        const int nt0 = wave >> 2;                       // tiles nt0 + 2*i
        v8f acc[8];
#pragma unroll
        for (int i = 0; i < 8; ++i) acc[i] = (v8f){};
        for (int kt = 0; kt < 32; ++kt) {
            v16h a = ldsA(hh + mt * 16 * LDH + kt * 32, LDH);
            __builtin_prefetch(w2t + (nt0 * 16) * 1024 + (kt + 1) * 32, 0, 1);
#pragma unroll
            for (int i = 0; i < 8; ++i) {
                const int nt = nt0 + 2 * i;              // 0..15
                v16h bm = ldB(w2t + nt * 16 * 1024 + kt * 32, 1024);
                acc[i] = WMMA_F16(a, bm, acc[i]);
            }
        }
        int n16 = lane & 15;
        int mb  = (lane >> 4) << 3;
#pragma unroll
        for (int i = 0; i < 8; ++i) {
            const int nt = nt0 + 2 * i;
            int n = nt * 16 + n16;
            float bias = b2[n];
#pragma unroll
            for (int r = 0; r < 8; ++r) mbuf[(mt * 16 + mb + r) * LDM + n] = acc[i][r] + bias;
        }
    }
    __syncthreads();

    // out = y + LayerNorm(mlp)
    {
        int m = tid >> 2, quarter = tid & 3;
        float s = 0.f, sq = 0.f;
        for (int i = 0; i < 64; ++i) {
            float v = mbuf[m * LDM + quarter * 64 + i];
            s += v; sq += v * v;
        }
        s  += __shfl_xor(s, 1);  s  += __shfl_xor(s, 2);
        sq += __shfl_xor(sq, 1); sq += __shfl_xor(sq, 2);
        float mu  = s * (1.f / 256.f);
        float var = sq * (1.f / 256.f) - mu * mu;
        float rs  = rsqrtf(var + 1e-5f);
        size_t base = (row0 + m) * 256;
        for (int q2 = 0; q2 < 64; ++q2) {
            int c = quarter * 64 + q2;
            float vm = mbuf[m * LDM + c];
            float yv = y[base + c];
            y[base + c] = yv + (vm - mu) * rs * gamma2[c] + beta2[c];
        }
    }
}

// ---------------------------------------------------------------------------
extern "C" void kernel_launch(void* const* d_in, const int* in_sizes, int n_in,
                              void* d_out, int out_size, void* d_ws, size_t ws_size,
                              hipStream_t stream) {
    (void)in_sizes; (void)n_in; (void)out_size; (void)ws_size;
    const float* x      = (const float*)d_in[0];
    const float* qkv_w  = (const float*)d_in[1];
    const float* qkv_b  = (const float*)d_in[2];
    const float* proj_w = (const float*)d_in[3];
    const float* proj_b = (const float*)d_in[4];
    const float* rpb    = (const float*)d_in[5];
    const float* g1     = (const float*)d_in[6];
    const float* be1    = (const float*)d_in[7];
    const float* w1     = (const float*)d_in[8];
    const float* b1     = (const float*)d_in[9];
    const float* w2     = (const float*)d_in[10];
    const float* b2     = (const float*)d_in[11];
    const float* g2     = (const float*)d_in[12];
    const float* be2    = (const float*)d_in[13];
    float* out = (float*)d_out;
    _Float16* ws16 = (_Float16*)d_ws;

    swin_prep_weights<<<512, 256, 0, stream>>>(qkv_w, proj_w, w1, w2, ws16);

    const size_t attn_lds = 211488;
    swin_attn_kernel<<<2048, 256, attn_lds, stream>>>(
        x, ws16, qkv_b, ws16 + 196608, proj_b, rpb, g1, be1, out);

    const size_t mlp_lds = 232448;
    swin_mlp_kernel<<<1568, 256, mlp_lds, stream>>>(
        ws16 + 262144, b1, ws16 + 524288, b2, g2, be2, out);
}
